// GraphEncoder_4037269258797
// MI455X (gfx1250) — matrix operand order repaired and checked by reading, compiled, not run on hardware
//
#include <hip/hip_runtime.h>
#include <hip/hip_bf16.h>

typedef _Float16 half16 __attribute__((ext_vector_type(16)));
typedef float    f32x8  __attribute__((ext_vector_type(8)));

#define NN   50000
#define EE   500000
#define GG   1024
#define HIDC 64
#define NHEAD 4
#define NLAYERS 3

// ---- monotonic float<->uint encoding for atomicMax on floats ----
__device__ __forceinline__ unsigned fenc(float f) {
  unsigned u = __float_as_uint(f);
  return (u & 0x80000000u) ? ~u : (u | 0x80000000u);
}
__device__ __forceinline__ float fdec(unsigned u) {
  unsigned v = (u & 0x80000000u) ? (u & 0x7fffffffu) : ~u;
  return __uint_as_float(v);
}

// =====================================================================
// WMMA GEMM: C[M,N] = A[M,K] @ B[K,N] (+bias), f32 mem, f16 WMMA compute.
// Block = 128 threads (4 waves). All 4 waves share one N-tile (16 cols)
// and cover 4 consecutive M-tiles. The K x 16 B tile is staged once into
// LDS, pre-swizzled into the CDNA5 WMMA B-operand layout (ISA 7.12.2,
// wave32) so each lane reads its 16 halves as one contiguous 32B vector:
//   B operand: lane l -> col n = l&15; half e -> K = kb*32 + (l>>4)*8 + (e&7) + (e>>3)*16
//   A operand: lane l -> row m = l&15; halves 0-7 -> K = kb*32 + (l>>4)*8 + e,
//              halves 8-15 -> +16
//   C/D:       vgpr v -> row = v + (l>>4)*8, col = l&15
// K is a compile-time constant -> K-loop fully unrolls into KB WMMAs.
// =====================================================================
template<int K>
__global__ void __launch_bounds__(128)
gemm_wmma_bias(const float* __restrict__ A, const float* __restrict__ B,
               const float* __restrict__ bias, float* __restrict__ C,
               int M, int N) {
  constexpr int KB = K / 32;
  __shared__ _Float16 lds_b[KB * 32 * 16];   // [kb][lane][halves 0..15]

  const int t    = threadIdx.x;
  const int lane = t & 31;
  const int wave = t >> 5;
  const int n0   = blockIdx.x * 16;

  // ---- cooperative stage: B[K x 16] -> LDS, swizzled to B-operand layout
#pragma unroll
  for (int i = 0; i < (K * 16) / 128; ++i) {
    int idx = i * 128 + t;
    int k = idx >> 4, n = idx & 15;
    float v = B[(size_t)k * N + n0 + n];
    int kb = k >> 5, r = k & 31;
    int l = (((r >> 3) & 1) << 4) | n;       // target lane = ((r>>3)&1)*16 + n
    int e = (r & 7) | ((r >> 4) << 3);
    lds_b[(kb * 32 + l) * 16 + e] = (_Float16)v;
  }
  __syncthreads();

  const int mt = blockIdx.y * 4 + wave;      // this wave's M tile
  if (mt * 16 >= M) return;                  // whole-wave skip (EXEC stays ~0)

  const int row = lane & 15;
  const int hi  = lane >> 4;

  f32x8 acc = {};
#pragma unroll
  for (int kb = 0; kb < KB; ++kb) {
    const int k0 = kb * 32;
    half16 a;
    const float* arow = A + (size_t)(mt * 16 + row) * K + k0 + hi * 8;
    __builtin_prefetch(arow + 32, 0, 0);     // global_prefetch_b8
#pragma unroll
    for (int e = 0; e < 8; ++e) {
      a[e]     = (_Float16)arow[e];          // K = k0 + hi*8 + e
      a[e + 8] = (_Float16)arow[e + 16];     // K = k0 + 16 + hi*8 + e
    }
    const half16 b = *(const half16*)&lds_b[(kb * 32 + lane) * 16];
    acc = __builtin_amdgcn_wmma_f32_16x16x32_f16(false, a, false, b,
                                                 (short)0, acc, false, false);
  }

  const int col = n0 + row;
  const float bz = bias ? bias[col] : 0.0f;
#pragma unroll
  for (int v = 0; v < 8; ++v) {
    int m = mt * 16 + v + hi * 8;
    C[(size_t)m * N + col] = acc[v] + bz;
  }
}

// ---------------------------------------------------------------------
__global__ void zero_u32(unsigned* __restrict__ p, size_t n) {
  size_t i = (size_t)blockIdx.x * blockDim.x + threadIdx.x;
  size_t st = (size_t)gridDim.x * blockDim.x;
  for (; i < n; i += st) p[i] = 0u;
}

// self-loop edge attr accumulation: deg[d] += 1, loop_attr[d] += attr[e]
__global__ void loop_attr_acc(const int* __restrict__ dst,
                              const float* __restrict__ edge_attr,
                              float* __restrict__ loop_attr,
                              float* __restrict__ deg) {
  int e = blockIdx.x * blockDim.x + threadIdx.x;
  if (e >= EE) return;
  int d = dst[e];
  atomicAdd(&deg[d], 1.0f);
  const float* a = edge_attr + (size_t)e * 8;
#pragma unroll
  for (int j = 0; j < 8; ++j) atomicAdd(&loop_attr[(size_t)d * 8 + j], a[j]);
}

__global__ void loop_attr_fin(float* __restrict__ loop_attr,
                              const float* __restrict__ deg) {
  int t = blockIdx.x * blockDim.x + threadIdx.x;
  if (t >= NN * 8) return;
  float dg = deg[t >> 3];
  loop_attr[t] /= (dg < 1.0f ? 1.0f : dg);
}

// M[l][d][h] = sum_c lin_edge_W[l][d][h*64+c] * att_edge[l][h][c]
__global__ void build_medge(const float* __restrict__ lin_edge_W,
                            const float* __restrict__ att_edge,
                            float* __restrict__ Medge) {
  int t = threadIdx.x;
  if (t >= NLAYERS * 32) return;
  int l = t >> 5, r = t & 31, d = r >> 2, h = r & 3;
  const float* W = lin_edge_W + (size_t)l * 8 * 256 + (size_t)d * 256 + h * 64;
  const float* a = att_edge + (size_t)l * 256 + h * 64;
  float s = 0.0f;
  for (int c = 0; c < 64; ++c) s += W[c] * a[c];
  Medge[t] = s;
}

// a_src[n,h], a_dst[n,h] from hs[n, h*64 + :]
__global__ void attn_coeff(const float* __restrict__ hs,
                           const float* __restrict__ att_src,
                           const float* __restrict__ att_dst,
                           float* __restrict__ a_src, float* __restrict__ a_dst) {
  int t = blockIdx.x * blockDim.x + threadIdx.x;
  if (t >= NN * NHEAD) return;
  int n = t >> 2, h = t & 3;
  const float* row = hs + (size_t)n * 256 + h * 64;
  const float* ws = att_src + h * 64;
  const float* wd = att_dst + h * 64;
  float s = 0.0f, d = 0.0f;
  for (int c = 0; c < 64; ++c) { float v = row[c]; s += v * ws[c]; d += v * wd[c]; }
  a_src[t] = s; a_dst[t] = d;
}

// pass 1: alpha = leaky_relu(a_src[s]+a_dst[d]+a_edge), segment-max to amax
__global__ void edge_alpha1(const int* __restrict__ src, const int* __restrict__ dst,
                            const float* __restrict__ edge_attr,
                            const float* __restrict__ loop_attr,
                            const float* __restrict__ Ml,
                            const float* __restrict__ a_src,
                            const float* __restrict__ a_dst,
                            float* __restrict__ alpha, unsigned* __restrict__ amax) {
  int e = blockIdx.x * blockDim.x + threadIdx.x;
  if (e >= EE + NN) return;
  int s, d; const float* attr;
  if (e < EE) { s = src[e]; d = dst[e]; attr = edge_attr + (size_t)e * 8; }
  else        { s = d = e - EE;         attr = loop_attr + (size_t)(e - EE) * 8; }
  float at[8];
#pragma unroll
  for (int j = 0; j < 8; ++j) at[j] = attr[j];
#pragma unroll
  for (int h = 0; h < NHEAD; ++h) {
    float ae = 0.0f;
#pragma unroll
    for (int j = 0; j < 8; ++j) ae += at[j] * Ml[j * 4 + h];
    float al = a_src[s * 4 + h] + a_dst[d * 4 + h] + ae;
    al = al > 0.0f ? al : 0.2f * al;          // leaky_relu(0.2)
    alpha[(size_t)e * 4 + h] = al;
    atomicMax(&amax[d * 4 + h], fenc(al));
  }
}

// pass 2: alpha = exp(alpha - amax[d]); segment-sum to asum
__global__ void edge_alpha2(const int* __restrict__ dst,
                            float* __restrict__ alpha,
                            const unsigned* __restrict__ amax,
                            float* __restrict__ asum) {
  int e = blockIdx.x * blockDim.x + threadIdx.x;
  if (e >= EE + NN) return;
  int d = (e < EE) ? dst[e] : e - EE;
#pragma unroll
  for (int h = 0; h < NHEAD; ++h) {
    float a = __expf(alpha[(size_t)e * 4 + h] - fdec(amax[d * 4 + h]));
    alpha[(size_t)e * 4 + h] = a;
    atomicAdd(&asum[d * 4 + h], a);
  }
}

// pass 3: agg[d] += (alpha/asum[d]) * hs[s]; one thread per (edge, 4 channels)
__global__ void edge_scatter(const int* __restrict__ src, const int* __restrict__ dst,
                             const float* __restrict__ alpha,
                             const float* __restrict__ asum,
                             const float* __restrict__ hs, float* __restrict__ agg) {
  long long tid = (long long)blockIdx.x * blockDim.x + threadIdx.x;
  const long long total = (long long)(EE + NN) * 64;
  if (tid >= total) return;
  int e = (int)(tid >> 6);
  int g = (int)(tid & 63);
  int h = g >> 4, c4 = (g & 15) * 4;
  int s, d;
  if (e < EE) { s = src[e]; d = dst[e]; } else { s = d = e - EE; }
  float w = alpha[(size_t)e * 4 + h] / (asum[d * 4 + h] + 1e-16f);
  const float4 v = *(const float4*)(hs + (size_t)s * 256 + h * 64 + c4);
  float* ag = agg + (size_t)d * 256 + h * 64 + c4;
  atomicAdd(ag + 0, w * v.x);
  atomicAdd(ag + 1, w * v.y);
  atomicAdd(ag + 2, w * v.z);
  atomicAdd(ag + 3, w * v.w);
}

// mean over heads + conv_bias, BN(eval), ReLU, residual
__global__ void node_update(const float* __restrict__ agg,
                            const float* __restrict__ conv_bias,
                            const float* __restrict__ gamma,
                            const float* __restrict__ beta,
                            const float* __restrict__ mean,
                            const float* __restrict__ var,
                            float* __restrict__ h) {
  int t = blockIdx.x * blockDim.x + threadIdx.x;
  if (t >= NN * HIDC) return;
  int n = t >> 6, c = t & 63;
  const float* a = agg + (size_t)n * 256 + c;
  float x = (a[0] + a[64] + a[128] + a[192]) * 0.25f + conv_bias[c];
  x = (x - mean[c]) * (gamma[c] * rsqrtf(var[c] + 1e-5f)) + beta[c];
  h[t] += (x > 0.0f ? x : 0.0f);
}

// global mean pool: accumulate
__global__ void pool_acc(const float* __restrict__ hout, const int* __restrict__ batch,
                         float* __restrict__ pool, float* __restrict__ cnt) {
  int t = blockIdx.x * blockDim.x + threadIdx.x;
  if (t >= NN * HIDC) return;
  int n = t >> 6, c = t & 63;
  int b = batch[n];
  if (c == 0) atomicAdd(&cnt[b], 1.0f);
  atomicAdd(&pool[(size_t)b * 64 + c], hout[t]);
}

__global__ void pool_fin(const float* __restrict__ pool, const float* __restrict__ cnt,
                         float* __restrict__ out) {
  int t = blockIdx.x * blockDim.x + threadIdx.x;
  if (t >= GG * HIDC) return;
  float c = cnt[t >> 6];
  out[t] = pool[t] / (c < 1.0f ? 1.0f : c);
}

// =====================================================================
extern "C" void kernel_launch(void* const* d_in, const int* in_sizes, int n_in,
                              void* d_out, int out_size, void* d_ws, size_t ws_size,
                              hipStream_t stream) {
  const float* x          = (const float*)d_in[0];
  const int*   edge_index = (const int*)d_in[1];
  const float* edge_attr  = (const float*)d_in[2];
  const int*   batch      = (const int*)d_in[3];
  const float* enc_W      = (const float*)d_in[4];
  const float* enc_b      = (const float*)d_in[5];
  const float* lin_W      = (const float*)d_in[6];
  const float* att_src    = (const float*)d_in[7];
  const float* att_dst    = (const float*)d_in[8];
  const float* lin_edge_W = (const float*)d_in[9];
  const float* att_edge   = (const float*)d_in[10];
  const float* conv_bias  = (const float*)d_in[11];
  const float* bn_gamma   = (const float*)d_in[12];
  const float* bn_beta    = (const float*)d_in[13];
  const float* bn_mean    = (const float*)d_in[14];
  const float* bn_var     = (const float*)d_in[15];
  const float* out_W      = (const float*)d_in[16];
  const float* out_b      = (const float*)d_in[17];
  float* out = (float*)d_out;

  const int* src = edge_index;
  const int* dst = edge_index + EE;

  // ---- workspace carve-out ----
  char* base = (char*)d_ws;
  size_t off = 0;
  auto alloc = [&](size_t bytes) -> void* {
    void* p = base + off;
    off += (bytes + 255) & ~(size_t)255;
    return p;
  };
  float*    h_buf     = (float*)alloc((size_t)NN * 64 * 4);
  float*    hs        = (float*)alloc((size_t)NN * 256 * 4);
  float*    agg       = (float*)alloc((size_t)NN * 256 * 4);
  float*    a_src_b   = (float*)alloc((size_t)NN * 4 * 4);
  float*    a_dst_b   = (float*)alloc((size_t)NN * 4 * 4);
  unsigned* amax      = (unsigned*)alloc((size_t)NN * 4 * 4);
  float*    asum      = (float*)alloc((size_t)NN * 4 * 4);
  float*    alpha     = (float*)alloc((size_t)(EE + NN) * 4 * 4);
  float*    loop_attr = (float*)alloc((size_t)NN * 8 * 4);
  float*    deg       = (float*)alloc((size_t)NN * 4);
  float*    Medge     = (float*)alloc((size_t)NLAYERS * 32 * 4);
  float*    pool      = (float*)alloc((size_t)GG * 64 * 4);
  float*    cnt       = (float*)alloc((size_t)GG * 4);
  float*    hout      = hs;  // reuse (hs free after last layer)

  const dim3 B128(128), B256(256);
  const int MB = (NN / 16 + 3) / 4;          // 782 M-blocks of 4 waves

  // 1) node encoder: h = x @ enc_W + enc_b  (K=128, N=64)
  gemm_wmma_bias<128><<<dim3(64 / 16, MB), B128, 0, stream>>>(x, enc_W, enc_b, h_buf, NN, 64);

  // 2) self-loop attrs (layer-independent)
  zero_u32<<<2048, B256, 0, stream>>>((unsigned*)loop_attr, (size_t)NN * 8);
  zero_u32<<<512,  B256, 0, stream>>>((unsigned*)deg, (size_t)NN);
  loop_attr_acc<<<(EE + 255) / 256, B256, 0, stream>>>(dst, edge_attr, loop_attr, deg);
  loop_attr_fin<<<(NN * 8 + 255) / 256, B256, 0, stream>>>(loop_attr, deg);

  // 3) collapse edge-attention weights to per-layer 8x4 matrices
  build_medge<<<1, 128, 0, stream>>>(lin_edge_W, att_edge, Medge);

  // 4) GAT layers
  for (int l = 0; l < NLAYERS; ++l) {
    gemm_wmma_bias<64><<<dim3(256 / 16, MB), B128, 0, stream>>>(
        h_buf, lin_W + (size_t)l * 64 * 256, nullptr, hs, NN, 256);
    attn_coeff<<<(NN * 4 + 255) / 256, B256, 0, stream>>>(
        hs, att_src + (size_t)l * 256, att_dst + (size_t)l * 256, a_src_b, a_dst_b);
    zero_u32<<<512,  B256, 0, stream>>>(amax, (size_t)NN * 4);
    zero_u32<<<512,  B256, 0, stream>>>((unsigned*)asum, (size_t)NN * 4);
    zero_u32<<<4096, B256, 0, stream>>>((unsigned*)agg, (size_t)NN * 256);
    edge_alpha1<<<(EE + NN + 255) / 256, B256, 0, stream>>>(
        src, dst, edge_attr, loop_attr, Medge + l * 32, a_src_b, a_dst_b, alpha, amax);
    edge_alpha2<<<(EE + NN + 255) / 256, B256, 0, stream>>>(dst, alpha, amax, asum);
    {
      long long total = (long long)(EE + NN) * 64;
      int blocks = (int)((total + 255) / 256);
      edge_scatter<<<blocks, B256, 0, stream>>>(src, dst, alpha, asum, hs, agg);
    }
    node_update<<<(NN * 64 + 255) / 256, B256, 0, stream>>>(
        agg, conv_bias + (size_t)l * 64, bn_gamma + (size_t)l * 64,
        bn_beta + (size_t)l * 64, bn_mean + (size_t)l * 64,
        bn_var + (size_t)l * 64, h_buf);
  }

  // 5) output layer + global mean pool
  gemm_wmma_bias<64><<<dim3(64 / 16, MB), B128, 0, stream>>>(h_buf, out_W, out_b, hout, NN, 64);
  zero_u32<<<256, B256, 0, stream>>>((unsigned*)pool, (size_t)GG * 64);
  zero_u32<<<16,  B256, 0, stream>>>((unsigned*)cnt, (size_t)GG);
  pool_acc<<<(NN * 64 + 255) / 256, B256, 0, stream>>>(hout, batch, pool, cnt);
  pool_fin<<<(GG * 64 + 255) / 256, B256, 0, stream>>>(pool, cnt, out);
}
